// DenseOpticalFlow_77773267796177
// MI455X (gfx1250) — matrix operand order repaired and checked by reading, not compiled, not run on hardware
//
#include <hip/hip_runtime.h>
#include <math.h>

typedef __attribute__((ext_vector_type(16))) _Float16 v16h;
typedef __attribute__((ext_vector_type(8)))  float    v8f;

static constexpr int IMG_H = 512;
static constexpr int IMG_W = 512;
static constexpr int NB    = 4;
static constexpr int GG    = 128;   // anchor grid (512/4)
static constexpr int WINSZ = 15;
static constexpr int RAD   = 7;

// 1D Gaussian tap: sigma = WIN/6 = 2.5 -> 1/(2*sigma^2) = 0.08
__device__ __forceinline__ float gauss_tap(int d) {
    float t = (float)(d - RAD);
    return __expf(-t * t * 0.08f);
}
__device__ __forceinline__ float gauss_invsum() {
    float s = 0.f;
#pragma unroll
    for (int t = 0; t < WINSZ; ++t) s += gauss_tap(t);
    return 1.f / s;
}

// ---------------------------------------------------------------------------
// Kernel 1: gray + sobel features, L2-normalized over channels.
// block (16,16) computes a 16x16 tile via an 18x18 LDS gray tile.
// z = b*2 + frameIdx
// ---------------------------------------------------------------------------
__global__ void feat_kernel(const float* __restrict__ frame1,
                            const float* __restrict__ frame2,
                            float* __restrict__ f1n,
                            float* __restrict__ f2n) {
    __shared__ float gt[18][19];
    const int z  = blockIdx.z;
    const int b  = z >> 1;
    const float* frame = (z & 1) ? frame2 : frame1;
    float*       out   = (z & 1) ? f2n    : f1n;
    const int x0 = blockIdx.x * 16 - 1;
    const int y0 = blockIdx.y * 16 - 1;
    const int tid = threadIdx.y * 16 + threadIdx.x;
    const size_t fb = (size_t)b * 3 * IMG_H * IMG_W;
    const size_t cs = (size_t)IMG_H * IMG_W;

    for (int i = tid; i < 18 * 18; i += 256) {
        int ly = i / 18, lx = i % 18;
        int y = y0 + ly, x = x0 + lx;
        float g = 0.f;
        if (y >= 0 && y < IMG_H && x >= 0 && x < IMG_W) {
            size_t p = fb + (size_t)y * IMG_W + x;
            g = 0.299f * frame[p] + 0.587f * frame[p + cs] + 0.114f * frame[p + 2 * cs];
        }
        gt[ly][lx] = g;
    }
    __syncthreads();

    const int lx = threadIdx.x + 1, ly = threadIdx.y + 1;
    float g  = gt[ly][lx];
    float fx = (gt[ly-1][lx+1] + 2.f*gt[ly][lx+1] + gt[ly+1][lx+1])
             - (gt[ly-1][lx-1] + 2.f*gt[ly][lx-1] + gt[ly+1][lx-1]);
    float fy = (gt[ly+1][lx-1] + 2.f*gt[ly+1][lx] + gt[ly+1][lx+1])
             - (gt[ly-1][lx-1] + 2.f*gt[ly-1][lx] + gt[ly-1][lx+1]);
    float n  = fmaxf(sqrtf(g*g + fx*fx + fy*fy), 1e-12f);
    float inv = 1.f / n;
    const int y = y0 + ly, x = x0 + lx;
    size_t o = fb + (size_t)y * IMG_W + x;
    out[o]          = g  * inv;
    out[o + cs]     = fx * inv;
    out[o + 2 * cs] = fy * inv;
}

// ---------------------------------------------------------------------------
// Kernel 2: correlation over 15x15 window, dual argmax -> grid displacements.
// one thread per anchor; fully unrolled so colmax[] stays in registers.
// ---------------------------------------------------------------------------
__global__ void corr_kernel(const float* __restrict__ f1n,
                            const float* __restrict__ f2n,
                            float* __restrict__ grid) {
    const int idx = blockIdx.x * blockDim.x + threadIdx.x;
    if (idx >= NB * GG * GG) return;
    const int b   = idx >> 14;
    const int rem = idx & (GG * GG - 1);
    const int gy  = rem >> 7;
    const int gx  = rem & (GG - 1);
    const int y1  = gy * 4, x1 = gx * 4;
    const size_t cs = (size_t)IMG_H * IMG_W;
    const size_t fb = (size_t)b * 3 * cs;

    size_t a = fb + (size_t)y1 * IMG_W + x1;
    const float a0 = f1n[a];
    const float a1 = f1n[a + cs];
    const float a2 = f1n[a + 2 * cs];

    const float NEG = -3.4028235e38f;
    float best = NEG; int bjy = 0;
    float colmax[WINSZ];
#pragma unroll
    for (int j = 0; j < WINSZ; ++j) colmax[j] = NEG;

    for (int jy = 0; jy < WINSZ; ++jy) {
        const int y2 = y1 + jy - RAD;
        const bool vy = (y2 >= 0) && (y2 < IMG_H);
        float rowmax = NEG;
#pragma unroll
        for (int jx = 0; jx < WINSZ; ++jx) {
            const int x2 = x1 + jx - RAD;
            float c = NEG;
            if (vy && x2 >= 0 && x2 < IMG_W) {
                size_t p = fb + (size_t)y2 * IMG_W + x2;
                c = a0 * f2n[p] + a1 * f2n[p + cs] + a2 * f2n[p + 2 * cs];
            }
            rowmax = fmaxf(rowmax, c);
            colmax[jx] = fmaxf(colmax[jx], c);
        }
        if (rowmax > best) { best = rowmax; bjy = jy; }   // strict > keeps first tie
    }
    float cb = NEG; int bjx = 0;
#pragma unroll
    for (int jx = 0; jx < WINSZ; ++jx)
        if (colmax[jx] > cb) { cb = colmax[jx]; bjx = jx; }

    const size_t go = ((size_t)(b * 2) * GG + gy) * GG + gx;
    grid[go]                     = (float)(bjx - RAD) / (float)IMG_W;  // disp_x (ch 0)
    grid[go + (size_t)GG * GG]   = (float)(bjy - RAD) / (float)IMG_H;  // disp_y (ch 1)
}

// ---------------------------------------------------------------------------
// Kernel 3: horizontal Gaussian 15-tap via WMMA.
// One wave per 16x16 tile. D(16y x 16x) = A(16x32 input slab) x B(32x16 Toeplitz).
// Input flow is grid upsampled 4x -> read grid[y>>2, x>>2] directly.
// ---------------------------------------------------------------------------
__global__ void hsmooth_kernel(const float* __restrict__ grid,
                               float* __restrict__ tmp) {
    const int z  = blockIdx.z;
    const int b  = z >> 1;
    const int ch = z & 1;
    const int x0 = blockIdx.x * 16;
    const int y0 = blockIdx.y * 16;
    const int lane = threadIdx.x;
    const int m  = lane & 15;
    const int hi = lane >> 4;
    const float invsum = gauss_invsum();

    const float* gch = grid + (size_t)(b * 2 + ch) * GG * GG;
    const int gy = (y0 + m) >> 2;

    // A fragment: 16x32 f16 (rows = y in tile, K = slab column)
    v16h A;
#pragma unroll
    for (int h = 0; h < 16; ++h) {
        const int K = hi * 16 + h;
        const int x = x0 - RAD + K;
        float v = 0.f;
        if (x >= 0 && x < IMG_W) v = gch[(size_t)gy * GG + (x >> 2)];
        A[h] = (_Float16)v;
    }
    // B fragment: 32x16 banded Toeplitz, B[k,n] = g[k-n]
    const int n = lane & 15;
    v16h Bt;
#pragma unroll
    for (int h = 0; h < 16; ++h) {
        const int d = hi * 16 + h - n;
        const float w = (d >= 0 && d < WINSZ) ? gauss_tap(d) * invsum : 0.f;
        Bt[h] = (_Float16)w;
    }
    v8f C = {};
    C = __builtin_amdgcn_wmma_f32_16x16x32_f16(false, A, false, Bt, (short)0, C, false, false);

    float* t = tmp + (size_t)(b * 2 + ch) * IMG_H * IMG_W;
#pragma unroll
    for (int r = 0; r < 8; ++r) {
        const int y = y0 + r + 8 * hi;   // D: VGPR r -> M=r (lanes 0-15), M=r+8 (lanes 16-31)
        t[(size_t)y * IMG_W + x0 + n] = C[r];
    }
}

// ---------------------------------------------------------------------------
// Kernel 4: vertical Gaussian via WMMA (both channels) + flow normalization.
// D(16y_out x 16x) = A(16x32 Toeplitz) x B(32x16 input slab).
// ---------------------------------------------------------------------------
__global__ void vsmooth_norm_kernel(const float* __restrict__ tmp,
                                    float* __restrict__ out) {
    const int b  = blockIdx.z;
    const int x0 = blockIdx.x * 16;
    const int y0 = blockIdx.y * 16;
    const int lane = threadIdx.x;
    const int m  = lane & 15;
    const int hi = lane >> 4;
    const int n  = m;
    const float invsum = gauss_invsum();

    // A: 16x32 Toeplitz, A[m,k] = g[k-m]
    v16h A;
#pragma unroll
    for (int h = 0; h < 16; ++h) {
        const int K = hi * 16 + h;
        const int d = K - m;
        const float w = (d >= 0 && d < WINSZ) ? gauss_tap(d) * invsum : 0.f;
        A[h] = (_Float16)w;
    }

    v8f D0, D1;
#pragma unroll
    for (int ch = 0; ch < 2; ++ch) {
        const float* t = tmp + (size_t)(b * 2 + ch) * IMG_H * IMG_W;
        v16h Bt;
#pragma unroll
        for (int h = 0; h < 16; ++h) {
            const int K = hi * 16 + h;
            const int y = y0 - RAD + K;
            float v = 0.f;
            if (y >= 0 && y < IMG_H) v = t[(size_t)y * IMG_W + x0 + n];
            Bt[h] = (_Float16)v;
        }
        v8f C = {};
        C = __builtin_amdgcn_wmma_f32_16x16x32_f16(false, A, false, Bt, (short)0, C, false, false);
        if (ch == 0) D0 = C; else D1 = C;
    }

#pragma unroll
    for (int r = 0; r < 8; ++r) {
        const float u = D0[r];
        const float v = D1[r];
        const float mag = fmaxf(sqrtf(u * u + v * v), 1e-6f);
        const float s = 1.f / (mag + 1e-6f);
        const int y = y0 + r + 8 * hi;
        const size_t o = ((size_t)(b * 2) * IMG_H + y) * IMG_W + x0 + n;
        out[o]                               = u * s;
        out[o + (size_t)IMG_H * IMG_W]       = v * s;
    }
}

// ---------------------------------------------------------------------------
extern "C" void kernel_launch(void* const* d_in, const int* in_sizes, int n_in,
                              void* d_out, int out_size, void* d_ws, size_t ws_size,
                              hipStream_t stream) {
    const float* frame1 = (const float*)d_in[0];
    const float* frame2 = (const float*)d_in[1];
    float* out = (float*)d_out;

    // workspace layout (~33 MB total)
    float* f1n  = (float*)d_ws;
    float* f2n  = f1n  + (size_t)NB * 3 * IMG_H * IMG_W;
    float* grid = f2n  + (size_t)NB * 3 * IMG_H * IMG_W;
    float* tmp  = grid + (size_t)NB * 2 * GG * GG;

    dim3 fblk(16, 16);
    dim3 fgrd(IMG_W / 16, IMG_H / 16, NB * 2);
    feat_kernel<<<fgrd, fblk, 0, stream>>>(frame1, frame2, f1n, f2n);

    const int nanchor = NB * GG * GG;
    corr_kernel<<<(nanchor + 255) / 256, 256, 0, stream>>>(f1n, f2n, grid);

    dim3 sgrd(IMG_W / 16, IMG_H / 16, NB * 2);
    hsmooth_kernel<<<sgrd, 32, 0, stream>>>(grid, tmp);

    dim3 vgrd(IMG_W / 16, IMG_H / 16, NB);
    vsmooth_norm_kernel<<<vgrd, 32, 0, stream>>>(tmp, out);
}
